// CInterpolateExtractor_50852412785189
// MI455X (gfx1250) — compile-verified, matrix operand order checked by reading, not run on hardware
//
#include <hip/hip_runtime.h>

// ---------------------------------------------------------------------------
// CDNA5 (gfx1250, wave32) fused bilinear-gather + 3-layer MLP using
// v_wmma_f32_16x16x32_bf16.  One block = 64 points; latent/h1/h2 ping-pong
// in LDS (2 x 64KB); weights pre-transposed to bf16 [Nout][K] in d_ws.
//
// GEMM loop structure: K is the outer rolled loop; per K-step each wave
// loads 4 A-frags (LDS) + NPW B-frags (global/L2) and issues NPW*4 WMMAs,
// keeping all NPW*4 f32 accumulators register-resident (no hoisting, no
// scratch spills).
// ---------------------------------------------------------------------------

typedef __attribute__((ext_vector_type(16))) __bf16        v16bf;
typedef __attribute__((ext_vector_type(8)))  float         v8f;
typedef __attribute__((ext_vector_type(4)))  unsigned int  u32x4;

union FragBF {
    v16bf v;
    u32x4 u[2];
};

__device__ __forceinline__ unsigned short f2bf(float f) {
    unsigned int u = __float_as_uint(f);
    u += 0x7FFFu + ((u >> 16) & 1u);           // round-to-nearest-even
    return (unsigned short)(u >> 16);
}

#define TM 64          // points (rows) per block
#define KP 512         // padded K (450 -> 512)

// ---------------------------------------------------------------------------
// One GEMM layer: [TM x 512] (bf16, LDS) x Wt[Nout x 512] (bf16, global,
// K-contiguous == B column-major) -> +bias (+relu) -> LDS bf16 or global f32.
// NPW = n-tiles this wave owns (Nout/16/8 waves).
// ---------------------------------------------------------------------------
template<int NPW, bool RELU, bool TO_LDS>
__device__ __forceinline__ void gemm_layer(
    const unsigned short* __restrict__ inLds,   // [TM][KP]
    const unsigned short* __restrict__ Wt,      // [Nout][KP]
    const float*          __restrict__ bias,    // [Nout]
    unsigned short*       __restrict__ outLds,  // [TM][KP]  (TO_LDS)
    float*                __restrict__ outG,    // [TM][256] (!TO_LDS)
    int lane, int wave)
{
    const int n    = lane & 15;
    const int hi   = lane >> 4;
    const int koff = hi * 8;                    // ISA 16-bit A/B lane K-offset

    v8f acc[NPW][4];
    #pragma unroll
    for (int j = 0; j < NPW; ++j)
        #pragma unroll
        for (int mt = 0; mt < 4; ++mt)
            acc[j][mt] = (v8f){0.f,0.f,0.f,0.f,0.f,0.f,0.f,0.f};

    const unsigned short* wptr[NPW];
    #pragma unroll
    for (int j = 0; j < NPW; ++j)
        wptr[j] = Wt + ((8 * j + wave) * 16 + n) * KP + koff;

    const unsigned short* aptr = inLds + n * KP + koff;

    #pragma clang loop unroll(disable)
    for (int kb = 0; kb < KP; kb += 32) {
        FragBF afr[4];                                   // A: 16x32 bf16 x4 M-tiles
        #pragma unroll
        for (int mt = 0; mt < 4; ++mt) {
            const u32x4* pA = (const u32x4*)(aptr + mt * 16 * KP + kb);
            afr[mt].u[0] = pA[0];                        // K = kb+koff+0..7
            afr[mt].u[1] = pA[2];                        // K = kb+koff+16..23
        }
        #pragma unroll
        for (int j = 0; j < NPW; ++j) {
            FragBF bfr;                                  // B: 32x16 bf16
            const u32x4* q = (const u32x4*)(wptr[j] + kb);
            bfr.u[0] = q[0];
            bfr.u[1] = q[2];
            #pragma unroll
            for (int mt = 0; mt < 4; ++mt)
                acc[j][mt] = __builtin_amdgcn_wmma_f32_16x16x32_bf16(
                    false, afr[mt].v, false, bfr.v, (short)0,
                    acc[j][mt], false, false);
        }
    }

    #pragma unroll
    for (int j = 0; j < NPW; ++j) {
        const int nt = 8 * j + wave;
        const float bv = bias[nt * 16 + n];
        #pragma unroll
        for (int mt = 0; mt < 4; ++mt) {
            #pragma unroll
            for (int r = 0; r < 8; ++r) {
                float v = acc[j][mt][r] + bv;
                if (RELU) v = fmaxf(v, 0.f);
                const int row = mt * 16 + r + 8 * hi;    // C/D row mapping
                if (TO_LDS) outLds[row * KP + nt * 16 + n] = f2bf(v);
                else        outG [row * 256 + nt * 16 + n] = v;
            }
        }
    }
}

// ---------------------------------------------------------------------------
// Fused kernel: gather 64 points' bilinear features into LDS latent (bf16,
// K-padded), then run the 3-layer MLP with WMMA.
// ---------------------------------------------------------------------------
__global__ __launch_bounds__(256)
void fused_sample_mlp(
    const float* __restrict__ f0, const float* __restrict__ f1,
    const float* __restrict__ f2, const float* __restrict__ pos,
    const unsigned short* __restrict__ Wt1, const float* __restrict__ b1,
    const unsigned short* __restrict__ Wt2, const float* __restrict__ b2,
    const unsigned short* __restrict__ Wt3, const float* __restrict__ b3,
    float* __restrict__ out)
{
    __shared__ unsigned short bufA[TM * KP];   // latent -> h2
    __shared__ unsigned short bufB[TM * KP];   // h1 (also gather scratch)

    const int tid       = threadIdx.x;
    const int lane      = tid & 31;
    const int wave      = tid >> 5;
    const int blockBase = blockIdx.x * TM;

    // gather scratch aliased on bufB (unused until layer-1 output)
    int*   sBase = (int*)bufB;                 // [3][TM]
    float* sWy   = (float*)bufB + 3 * TM;      // [3][TM]
    float* sWx   = sWy + 3 * TM;               // [3][TM]

    // ---- phase 1: per-point corner bases + weights, pos cols, zero pad ----
    if (tid < TM) {
        const int p = blockBase + tid;
        const int b = p >> 13;                 // N = 8192
        const float py = pos[2 * p + 0];
        const float px = pos[2 * p + 1];
        bufA[tid * KP + 0] = f2bf(py);
        bufA[tid * KP + 1] = f2bf(px);
        const int Hs[3] = {128, 64, 32}, Cs[3] = {64, 128, 256};
        #pragma unroll
        for (int l = 0; l < 3; ++l) {
            const int H = Hs[l], W = Hs[l], C = Cs[l];
            float y = py * (float)(H - 1), x = px * (float)(W - 1);
            float y0 = fminf(fmaxf(floorf(y), 0.f), (float)(H - 2));
            float x0 = fminf(fmaxf(floorf(x), 0.f), (float)(W - 2));
            sBase[l * TM + tid] = ((b * H + (int)y0) * W + (int)x0) * C;
            sWy[l * TM + tid] = y - y0;
            sWx[l * TM + tid] = x - x0;
        }
    }
    for (int idx = tid; idx < TM * 64; idx += 256) {   // zero cols 450..511
        const int i = idx >> 6;
        const int c = (idx & 63) + 448;
        if (c >= 450) bufA[i * KP + c] = 0;
    }
    __syncthreads();

    // ---- phase 2: bilinear channel fill (coalesced along C) ----
    auto fill = [&](const float* __restrict__ f, int lvl, int C, int csh, int off) {
        const int rowStride = 8192;            // W*C == 8192 floats all levels
        for (int e = tid; e < TM * C; e += 256) {
            const int i = e >> csh, c = e & (C - 1);
            const float* fp = f + sBase[lvl * TM + i] + c;
            const float wy = sWy[lvl * TM + i], wx = sWx[lvl * TM + i];
            const float v00 = fp[0], v01 = fp[C];
            const float v10 = fp[rowStride], v11 = fp[rowStride + C];
            const float top = fmaf(wx, v01 - v00, v00);
            const float bot = fmaf(wx, v11 - v10, v10);
            bufA[i * KP + off + c] = f2bf(fmaf(wy, bot - top, top));
        }
    };
    fill(f0, 0,  64, 6,   2);
    fill(f1, 1, 128, 7,  66);
    fill(f2, 2, 256, 8, 194);
    __syncthreads();

    // ---- phase 3: MLP, ping-pong LDS ----
    gemm_layer<4, true,  true >(bufA, Wt1, b1, bufB, nullptr, lane, wave);
    __syncthreads();
    gemm_layer<4, true,  true >(bufB, Wt2, b2, bufA, nullptr, lane, wave);
    __syncthreads();
    gemm_layer<2, false, false>(bufA, Wt3, b3, nullptr,
                                out + (size_t)blockBase * 256, lane, wave);
}

// ---------------------------------------------------------------------------
// Weight prep: transpose + bf16-convert (+K-pad W1) into ws.
// ---------------------------------------------------------------------------
__global__ __launch_bounds__(256)
void prep_weights(const float* __restrict__ W1, const float* __restrict__ W2,
                  const float* __restrict__ W3,
                  unsigned short* __restrict__ Wt1,
                  unsigned short* __restrict__ Wt2,
                  unsigned short* __restrict__ Wt3)
{
    const int t = blockIdx.x * 256 + threadIdx.x;
    if (t < 512 * 512) {
        const int n = t >> 9, k = t & 511;
        Wt1[n * 512 + k] = (k < 450) ? f2bf(W1[k * 512 + n]) : (unsigned short)0;
    } else if (t < 2 * 512 * 512) {
        const int u = t - 512 * 512;
        const int n = u >> 9, k = u & 511;
        Wt2[n * 512 + k] = f2bf(W2[k * 512 + n]);
    } else if (t < 2 * 512 * 512 + 256 * 512) {
        const int u = t - 2 * 512 * 512;
        const int n = u >> 9, k = u & 511;
        Wt3[n * 512 + k] = f2bf(W3[k * 256 + n]);
    }
}

extern "C" void kernel_launch(void* const* d_in, const int* in_sizes, int n_in,
                              void* d_out, int out_size, void* d_ws, size_t ws_size,
                              hipStream_t stream)
{
    const float* f0  = (const float*)d_in[0];
    const float* f1  = (const float*)d_in[1];
    const float* f2  = (const float*)d_in[2];
    const float* pos = (const float*)d_in[3];
    const float* W1  = (const float*)d_in[4];
    const float* b1  = (const float*)d_in[5];
    const float* W2  = (const float*)d_in[6];
    const float* b2  = (const float*)d_in[7];
    const float* W3  = (const float*)d_in[8];
    const float* b3  = (const float*)d_in[9];

    unsigned short* Wt1 = (unsigned short*)d_ws;       // 512x512 bf16
    unsigned short* Wt2 = Wt1 + 512 * 512;             // 512x512 bf16
    unsigned short* Wt3 = Wt2 + 512 * 512;             // 256x512 bf16

    prep_weights<<<2560, 256, 0, stream>>>(W1, W2, W3, Wt1, Wt2, Wt3);

    const int BN = 32 * 8192;
    fused_sample_mlp<<<BN / TM, 256, 0, stream>>>(
        f0, f1, f2, pos, Wt1, b1, Wt2, b2, Wt3, b3, (float*)d_out);
}
